// AttenConv_63909113364810
// MI455X (gfx1250) — matrix operand order repaired and checked by reading, not compiled, not run on hardware
//
#include <hip/hip_runtime.h>
#include <hip/hip_bf16.h>

typedef __attribute__((ext_vector_type(16))) _Float16 v16h;
typedef __attribute__((ext_vector_type(8)))  float    v8f;

#define HEADS 8
#define DHEAD 64
#define INNER 512           // HEADS * DHEAD
#define SEQ   2048
#define NB    4             // batch
#define QD    256
#define CD    256

// ---------------------------------------------------------------------------
// WMMA helper: D = A(16x32 f16) * B(32x16 f16) + C(16x16 f32)
// ---------------------------------------------------------------------------
__device__ __forceinline__ v8f wmma16(v16h a, v16h b, v8f c) {
  return __builtin_amdgcn_wmma_f32_16x16x32_f16(
      /*neg_a=*/false, a, /*neg_b=*/false, b,
      /*c_mod=*/(short)0, c, /*reuse_a=*/false, /*reuse_b=*/false);
}

// Async 16-byte global -> LDS copy (per lane). Tracked by ASYNCcnt.
__device__ __forceinline__ void async16(unsigned int ldsOff, const void* g) {
  asm volatile("global_load_async_to_lds_b128 %0, %1, off"
               :: "v"(ldsOff), "v"(g) : "memory");
}

// A-fragment (16x32, 16-bit) from row-major f32 matrix X[.., ld].
// This code's contiguous k order (not the ISA table): lane l (0..15) holds row M=l; lanes 16..31 same M, K offset +16.
// VGPR v holds the K pair at 2*v (+hi*16).
__device__ __forceinline__ v16h load_a_f32(const float* __restrict__ X, int ld,
                                           int row, int kBase, int hi) {
  const float* p = X + (size_t)row * ld + kBase + hi * 16;
  v16h a;
#pragma unroll
  for (int v = 0; v < 8; ++v) {
    int k0 = 2 * v;
    a[2 * v]     = (_Float16)p[k0];
    a[2 * v + 1] = (_Float16)p[k0 + 1];
  }
  return a;
}

// B-fragment (32x16) where B = W^T, W row-major f32 [Cout, ld]:
// B[k][n] = W[n][k]. Lane l holds column n=l; element j is K = kBase + hi*16 + j.
__device__ __forceinline__ v16h load_b_wT(const float* __restrict__ W, int ld,
                                          int n, int kBase, int hi) {
  const float* p = W + (size_t)n * ld + kBase + hi * 16;
  v16h b;
#pragma unroll
  for (int j = 0; j < 16; ++j) b[j] = (_Float16)p[j];
  return b;
}

// ---------------------------------------------------------------------------
// Kernel 1: projection GEMM  Y[R,Cout] = X[R,K] * W[Cout,K]^T + bias
// mode 0: Q    -> out0[b][h][s][d]                       (Cout=512)
// mode 1: K,V  -> out0[b][h][s][d], out1[b][h][d][s] (V TRANSPOSED; Cout=1024)
// ---------------------------------------------------------------------------
__global__ __launch_bounds__(128) void proj_kernel(
    const float* __restrict__ X, const float* __restrict__ W,
    const float* __restrict__ bias, int Kdim, int mode,
    _Float16* __restrict__ out0, _Float16* __restrict__ out1)
{
  int lane = threadIdx.x & 31;
  int l = lane & 15, hi = lane >> 4;
  int rowBase = blockIdx.x * 16;
  int nBase   = (blockIdx.y * 4 + threadIdx.y) * 16;

  v8f acc = {};
  for (int kb = 0; kb < Kdim; kb += 32) {
    v16h a = load_a_f32(X, Kdim, rowBase + l, kb, hi);
    v16h b = load_b_wT(W, Kdim, nBase + l, kb, hi);
    acc = wmma16(a, b, acc);
  }

  int col = nBase + l;
  float bv = bias[col];
#pragma unroll
  for (int r = 0; r < 8; ++r) {
    int row = rowBase + r + hi * 8;      // C-tile: VGPR r -> row r (+8 for hi lanes)
    float val = acc[r] + bv;
    int bb = row >> 11;                  // / 2048
    int s  = row & 2047;
    int e  = col;
    if (mode == 1 && e >= INNER) {
      // V: store transposed (b,h,d,s) so attention B-fragments are contiguous
      e -= INNER;
      int h = e >> 6, d = e & 63;
      out1[((size_t)(bb * HEADS + h) * DHEAD + d) * SEQ + s] = (_Float16)val;
    } else {
      int h = e >> 6, d = e & 63;
      out0[((size_t)(bb * HEADS + h) * SEQ + s) * DHEAD + d] = (_Float16)val;
    }
  }
}

// ---------------------------------------------------------------------------
// Kernel 2: fused flash attention.
// Block = 4 waves, all on the same (b,h); per iteration the block stages one
// 32-key K-block (4KB) + Vt-block (4KB) into LDS via async global->LDS copies
// (double-buffered, overlapped with compute). Each wave owns 32 query rows.
// ---------------------------------------------------------------------------
__global__ __launch_bounds__(128) void attn_kernel(
    const _Float16* __restrict__ Qh, const _Float16* __restrict__ Kh,
    const _Float16* __restrict__ Vt, _Float16* __restrict__ Oh)
{
  __shared__ _Float16 Kbuf[2][32 * DHEAD];   // K rows kb..kb+31   (2 x 4KB)
  __shared__ _Float16 Vbuf[2][DHEAD * 32];   // Vt rows d=0..63, 32 cols (2 x 4KB)
  __shared__ _Float16 Pbuf[4][16][32];       // per-wave P staging (1KB each)

  int lane = threadIdx.x & 31;
  int l = lane & 15, hi = lane >> 4;
  int tid = threadIdx.y * 32 + threadIdx.x;  // 0..127, for staging chunks
  int w  = blockIdx.x * 4 + threadIdx.y;     // global wave id, 2048 total
  int bh = w >> 6;                           // 64 32-query tiles per (b,h)
  int qt = w & 63;
  int b = bh >> 3, h = bh & 7;

  const _Float16* Qp = Qh + (size_t)(b * HEADS + h) * SEQ * DHEAD;
  const _Float16* Kp = Kh + (size_t)(b * HEADS + h) * SEQ * DHEAD;
  const _Float16* Vp = Vt + (size_t)(b * HEADS + h) * DHEAD * SEQ; // [d][s]

  unsigned int koff[2], voff[2];
  koff[0] = (unsigned int)(uintptr_t)&Kbuf[0][0];
  koff[1] = (unsigned int)(uintptr_t)&Kbuf[1][0];
  voff[0] = (unsigned int)(uintptr_t)&Vbuf[0][0];
  voff[1] = (unsigned int)(uintptr_t)&Vbuf[1][0];

  // Persistent Q A-fragments: qa[t][khalf], t = q-subtile (16 rows each)
  v16h qa[2][2];
#pragma unroll
  for (int t = 0; t < 2; ++t) {
    const _Float16* qrow = Qp + (size_t)(qt * 32 + t * 16 + l) * DHEAD;
#pragma unroll
    for (int v = 0; v < 8; ++v) {
      int k0 = 2 * v + hi * 16;
      qa[t][0][2 * v]     = qrow[k0];
      qa[t][0][2 * v + 1] = qrow[k0 + 1];
      qa[t][1][2 * v]     = qrow[32 + k0];
      qa[t][1][2 * v + 1] = qrow[32 + k0 + 1];
    }
  }

  v8f o[2][4];                 // O tiles: 32 rows x 64 dims, f32
  float mrow[2][8], lrow[2][8];
#pragma unroll
  for (int t = 0; t < 2; ++t) {
#pragma unroll
    for (int dt = 0; dt < 4; ++dt) o[t][dt] = (v8f){};
#pragma unroll
    for (int r = 0; r < 8; ++r) { mrow[t][r] = -1e30f; lrow[t][r] = 0.0f; }
  }

  const float scale = 0.125f;  // DIM_HEAD^-0.5
  _Float16 (*P)[32] = Pbuf[threadIdx.y];

  // Cooperative staging of one 32-key block: 512 chunks of 16B across 128
  // threads = 4 async instructions per thread (2 for K, 2 for Vt).
  auto stage = [&](int kb, int buf) {
    const char* kbase = (const char*)(Kp + (size_t)kb * DHEAD); // contiguous 4KB
#pragma unroll
    for (int i = 0; i < 2; ++i) {
      int c = tid + i * 128;                 // 16B chunk id, 0..255
      async16(koff[buf] + c * 16, kbase + c * 16);
      int d = c >> 2, ob = (c & 3) * 16;     // Vt row d, byte offset within 64B
      async16(voff[buf] + c * 16,
              (const char*)(Vp + (size_t)d * SEQ + kb) + ob);
    }
  };

  stage(0, 0);   // prologue: first block in flight

  for (int kb = 0; kb < SEQ; kb += 32) {
    int ib = (kb >> 5) & 1;
    if (kb + 32 < SEQ) {
      stage(kb + 32, ib ^ 1);                          // overlap next block
      asm volatile("s_wait_asynccnt 0x4" ::: "memory"); // retire current block only
    } else {
      asm volatile("s_wait_asynccnt 0x0" ::: "memory");
    }
    __syncthreads();   // current block visible to all 4 waves

    const _Float16* kc = &Kbuf[ib][0];
    const _Float16* vc = &Vbuf[ib][0];

    // ---- K B-fragments for 32 keys from LDS (ds_load_b128) ----
    v16h kb00 = *(const v16h*)(kc + (size_t)(l)      * DHEAD + hi * 16);
    v16h kb01 = *(const v16h*)(kc + (size_t)(l)      * DHEAD + 32 + hi * 16);
    v16h kb10 = *(const v16h*)(kc + (size_t)(16 + l) * DHEAD + hi * 16);
    v16h kb11 = *(const v16h*)(kc + (size_t)(16 + l) * DHEAD + 32 + hi * 16);

    v16h pa[2];
#pragma unroll
    for (int t = 0; t < 2; ++t) {
      // ---- S = Q K^T (two 16x16 tiles covering 32 keys) ----
      v8f s0 = {}, s1 = {};
      s0 = wmma16(qa[t][0], kb00, s0);
      s0 = wmma16(qa[t][1], kb01, s0);
      s1 = wmma16(qa[t][0], kb10, s1);
      s1 = wmma16(qa[t][1], kb11, s1);

      // ---- online softmax update (rows live across 16-lane halves) ----
#pragma unroll
      for (int r = 0; r < 8; ++r) {
        float x0 = s0[r] * scale, x1 = s1[r] * scale;
        float mx = fmaxf(x0, x1);
#pragma unroll
        for (int msk = 1; msk < 16; msk <<= 1)
          mx = fmaxf(mx, __shfl_xor(mx, msk, 16));
        float mnew  = fmaxf(mrow[t][r], mx);
        float alpha = __expf(mrow[t][r] - mnew);
        float p0 = __expf(x0 - mnew), p1 = __expf(x1 - mnew);
        float rs = p0 + p1;
#pragma unroll
        for (int msk = 1; msk < 16; msk <<= 1)
          rs += __shfl_xor(rs, msk, 16);
        lrow[t][r] = lrow[t][r] * alpha + rs;
        mrow[t][r] = mnew;
        o[t][0][r] *= alpha; o[t][1][r] *= alpha;
        o[t][2][r] *= alpha; o[t][3][r] *= alpha;
        s0[r] = p0; s1[r] = p1;
      }

      // ---- C-layout P -> LDS -> A-layout fragment (per-wave, in-order DS) ----
#pragma unroll
      for (int r = 0; r < 8; ++r) {
        P[r + hi * 8][l]      = (_Float16)s0[r];
        P[r + hi * 8][16 + l] = (_Float16)s1[r];
      }
      asm volatile("s_wait_dscnt 0" ::: "memory");
      const _Float16* prow = &P[l][hi * 16];
#pragma unroll
      for (int v = 0; v < 8; ++v) {
        int k0 = 2 * v;
        pa[t][2 * v]     = prow[k0];
        pa[t][2 * v + 1] = prow[k0 + 1];
      }
    }

    // ---- O += P * V from LDS: B[k][n] = Vbuf[(dt*16+n)*32 + k] ----
#pragma unroll
    for (int dt = 0; dt < 4; ++dt) {
      v16h bv = *(const v16h*)(vc + (size_t)(dt * 16 + l) * 32 + hi * 16);
      o[0][dt] = wmma16(pa[0], bv, o[0][dt]);
      o[1][dt] = wmma16(pa[1], bv, o[1][dt]);
    }

    __syncthreads();   // everyone done reading; buffer may be overwritten
  }

  // ---- epilogue: normalize and write O in (b, n, h*64+d) f16 layout ----
#pragma unroll
  for (int t = 0; t < 2; ++t) {
#pragma unroll
    for (int r = 0; r < 8; ++r) {
      float inv = 1.0f / lrow[t][r];
      int q = qt * 32 + t * 16 + r + hi * 8;
      _Float16* dst = Oh + ((size_t)(b * SEQ + q) * INNER) + h * DHEAD;
      dst[l]      = (_Float16)(o[t][0][r] * inv);
      dst[16 + l] = (_Float16)(o[t][1][r] * inv);
      dst[32 + l] = (_Float16)(o[t][2][r] * inv);
      dst[48 + l] = (_Float16)(o[t][3][r] * inv);
    }
  }
}

// ---------------------------------------------------------------------------
// Kernel 3: output projection  Y[8192,256] = O[8192,512](f16) * Wo[256,512]^T + bo
// ---------------------------------------------------------------------------
__global__ __launch_bounds__(128) void outproj_kernel(
    const _Float16* __restrict__ Oh, const float* __restrict__ Wo,
    const float* __restrict__ bo, float* __restrict__ Y)
{
  int lane = threadIdx.x & 31;
  int l = lane & 15, hi = lane >> 4;
  int rowBase = blockIdx.x * 16;
  int nBase   = (blockIdx.y * 4 + threadIdx.y) * 16;

  v8f acc = {};
  for (int kb = 0; kb < INNER; kb += 32) {
    const _Float16* arow = Oh + (size_t)(rowBase + l) * INNER + kb + hi * 16;
    v16h a;
#pragma unroll
    for (int v = 0; v < 8; ++v) {
      int k0 = 2 * v;
      a[2 * v]     = arow[k0];
      a[2 * v + 1] = arow[k0 + 1];
    }
    v16h b = load_b_wT(Wo, INNER, nBase + l, kb, hi);
    acc = wmma16(a, b, acc);
  }

  int col = nBase + l;
  float bv = bo[col];
#pragma unroll
  for (int r = 0; r < 8; ++r) {
    int row = rowBase + r + hi * 8;
    Y[(size_t)row * QD + col] = acc[r] + bv;
  }
}

// ---------------------------------------------------------------------------
extern "C" void kernel_launch(void* const* d_in, const int* in_sizes, int n_in,
                              void* d_out, int out_size, void* d_ws, size_t ws_size,
                              hipStream_t stream)
{
  const float* x   = (const float*)d_in[0];   // (4,2048,256)
  const float* ctx = (const float*)d_in[1];   // (4,2048,256)
  const float* Wq  = (const float*)d_in[2];   // (512,256)
  const float* bq  = (const float*)d_in[3];   // (512,)
  const float* Wkv = (const float*)d_in[4];   // (1024,256)
  const float* bkv = (const float*)d_in[5];   // (1024,)
  const float* Wo  = (const float*)d_in[6];   // (256,512)
  const float* bo  = (const float*)d_in[7];   // (256,)
  float* Y = (float*)d_out;                   // (4,2048,256)

  const size_t perTensor = (size_t)NB * HEADS * SEQ * DHEAD;  // 4 Mi halfs = 8 MB
  _Float16* Qh = (_Float16*)d_ws;
  _Float16* Kh = Qh + perTensor;
  _Float16* Vt = Kh + perTensor;              // (b,h,d,s) transposed V
  _Float16* Oh = Vt + perTensor;              // (b,n,inner)

  dim3 blk(32, 4);  // 4 waves per block

  // Q projection: R=8192, Cout=512
  proj_kernel<<<dim3(512, 512 / 64), blk, 0, stream>>>(x, Wq, bq, QD, 0, Qh, nullptr);
  // KV projection: R=8192, Cout=1024 (K into Kh, V transposed into Vt)
  proj_kernel<<<dim3(512, 1024 / 64), blk, 0, stream>>>(ctx, Wkv, bkv, CD, 1, Kh, Vt);
  // Fused attention: 2048 waves (32 queries each), 4 waves/block
  attn_kernel<<<dim3(512), blk, 0, stream>>>(Qh, Kh, Vt, Oh);
  // Output projection: R=8192, Cout=256
  outproj_kernel<<<dim3(512, 256 / 64), blk, 0, stream>>>(Oh, Wo, bo, Y);
}
